// PDFCNetlayer_28200755266007
// MI455X (gfx1250) — compile-verified, matrix operand order checked by reading, not run on hardware
//
#include <hip/hip_runtime.h>

// MI455X / gfx1250, wave32. WMMA f32<-f16 16x16x32 for both GEMM stages.
// fc_gemm: A tile staged by the Tensor Data Mover (tensor_load_to_lds),
//          B tile staged+transposed by threads; fragments = ds_load_b128.

typedef __attribute__((ext_vector_type(16))) _Float16 v16h;
typedef __attribute__((ext_vector_type(8)))  _Float16 v8h;
typedef __attribute__((ext_vector_type(4)))  _Float16 v4h;
typedef __attribute__((ext_vector_type(8)))  float    v8f;
typedef unsigned int u32x4 __attribute__((ext_vector_type(4)));
typedef int          i32x8 __attribute__((ext_vector_type(8)));
typedef int          i32x4 __attribute__((ext_vector_type(4)));

#define RR 3
#define NN 20000
#define EE 320000
#define FF 256
#define HH 8
#define OO 64
#define DD 512

// ---- workspace layout (units: 4-byte elements) ----
// [ z : R*N*D f32 ][ emax : R*N*H u32 keys ][ den : R*N*H f32 ][ wsum : 8 f32 ]
// [ h : N*D f32 ][ el : R*N*H ][ er : R*N*H ]      total ~171.5 MB
static constexpr size_t Z_CNT    = (size_t)RR * NN * DD;
static constexpr size_t RNH      = (size_t)RR * NN * HH;
static constexpr size_t EMAX_OFF = Z_CNT;
static constexpr size_t DEN_OFF  = EMAX_OFF + RNH;
static constexpr size_t WSUM_OFF = DEN_OFF + RNH;
static constexpr size_t H_OFF    = WSUM_OFF + 8;             // zeroed region ends here
static constexpr size_t EL_OFF   = H_OFF + (size_t)NN * DD;
static constexpr size_t ER_OFF   = EL_OFF + RNH;

// order-preserving float <-> uint key for atomicMax-based float max
__device__ __forceinline__ unsigned fkey(float v) {
    unsigned b = __float_as_uint(v);
    return (b & 0x80000000u) ? ~b : (b | 0x80000000u);
}
__device__ __forceinline__ float fdec(unsigned k) {
    unsigned b = (k & 0x80000000u) ? (k & 0x7fffffffu) : ~k;
    return __uint_as_float(b);
}

__device__ __forceinline__ v16h frag_cat(v8h lo, v8h hi) {
    return __builtin_shufflevector(lo, hi, 0, 1, 2, 3, 4, 5, 6, 7,
                                           8, 9, 10, 11, 12, 13, 14, 15);
}

// ---------------------------------------------------------------------------
// FC GEMM: C[M,Nc] = A[M,K] * B[K,Nc]  via v_wmma_f32_16x16x32_f16
// block = 256 threads = 8 waves (4 row-tiles x 2 col-tiles), tile 64x64, KC=32
// A tile: TDM 2D copy global->LDS (f32, padded row stride 36 DW for banks),
//         tensor_dim1 = M-rowBase so OOB rows read as zero (ragged tail).
// B tile: threads stage transposed f16 BsT[n][k]; fragments = 2x ds_load_b128.
// ---------------------------------------------------------------------------
__global__ __launch_bounds__(256)
void fc_gemm_f16wmma(const float* __restrict__ A, const float* __restrict__ B,
                     float* __restrict__ C, int M, int K, int Nc) {
    __shared__ __align__(16) float    Af32[64][36];   // TDM dest: 32 DW rows + 4 DW pad
    __shared__ __align__(16) _Float16 BsT[64][40];
    const int tid  = threadIdx.x;
    const int lane = tid & 31, wid = tid >> 5;
    const int rw   = wid & 3,  cw  = wid >> 2;
    const int half = lane >> 4, mrow = lane & 15;
    const int rowBase = blockIdx.x * 64;
    const int colBase = blockIdx.y * 64;
    const bool full = (rowBase + 64 <= M);
    const int h8  = half * 8;
    const int am  = rw * 16 + mrow;
    const int bn0 = cw * 32 + mrow, bn1 = bn0 + 16;
    v8f acc0 = {}, acc1 = {};

    // D# pieces that are loop-invariant
    const unsigned ldsA   = (unsigned)(uintptr_t)&Af32[0][0];
    const unsigned tdim1  = (unsigned)(M - rowBase);           // remaining rows (OOB -> 0)
    const unsigned long long gbase =
        (unsigned long long)(uintptr_t)(A + (size_t)rowBase * K);

    for (int kk = 0; kk < K; kk += 32) {
        // ---- A tile via Tensor Data Mover ----
        if (tid == 0) {
            unsigned long long ga = gbase + (unsigned long long)kk * 4ull;
            u32x4 g0;
            g0[0] = 1u;                                        // count=1
            g0[1] = ldsA;                                      // lds_addr
            g0[2] = (unsigned)(ga & 0xffffffffull);            // global_addr[31:0]
            g0[3] = (unsigned)((ga >> 32) & 0x01ffffffull)     // global_addr[56:32]
                  | 0x80000000u;                               // type=2 ("image")
            i32x8 g1;
            g1[0] = (int)((2u << 16)                           // data_size = 4B
                        | (1u << 20)                           // pad_enable
                        | (4u << 22)                           // pad_interval = 32 DW
                        | (3u << 25));                         // pad_amount   = 4 DW
            g1[1] = (int)((32u & 0xffffu) << 16);              // tensor_dim0 = 32
            g1[2] = (int)(((32u >> 16) & 0xffffu) | ((tdim1 & 0xffffu) << 16));
            g1[3] = (int)(((tdim1 >> 16) & 0xffffu) | (32u << 16));  // tile_dim0 = 32
            g1[4] = (int)(64u);                                // tile_dim1 = 64, tile_dim2 = 0
            g1[5] = (int)(unsigned)K;                          // tensor_dim0_stride lo
            g1[6] = 0;                                         // stride hi, dim1_stride lo
            g1[7] = 0;
            i32x4 gz  = {0, 0, 0, 0};
            i32x8 gz8 = {0, 0, 0, 0, 0, 0, 0, 0};
            __builtin_amdgcn_tensor_load_to_lds(g0, g1, gz, gz, gz8, 0);
            __builtin_amdgcn_s_wait_tensorcnt(0);
        }
        // ---- B tile 32x64 transposed (BsT[n][k]), float4 global reads ----
        #pragma unroll
        for (int t = 0; t < 2; ++t) {
            int idx = tid + t * 256;                 // 512 quads
            int rB = idx >> 4, c4 = (idx & 15) << 2;
            const float4 v = *(const float4*)(B + (size_t)(kk + rB) * Nc + colBase + c4);
            BsT[c4 + 0][rB] = (_Float16)v.x;
            BsT[c4 + 1][rB] = (_Float16)v.y;
            BsT[c4 + 2][rB] = (_Float16)v.z;
            BsT[c4 + 3][rB] = (_Float16)v.w;
        }
        __syncthreads();

        // A fragment: f32 from LDS (4x b128) -> packed cvt to f16
        const float* arow = &Af32[am][0];
        float4 q0 = *(const float4*)(arow + h8);
        float4 q1 = *(const float4*)(arow + h8 + 4);
        float4 q2 = *(const float4*)(arow + h8 + 16);
        float4 q3 = *(const float4*)(arow + h8 + 20);
        v16h a;
        a[0]  = (_Float16)q0.x; a[1]  = (_Float16)q0.y;
        a[2]  = (_Float16)q0.z; a[3]  = (_Float16)q0.w;
        a[4]  = (_Float16)q1.x; a[5]  = (_Float16)q1.y;
        a[6]  = (_Float16)q1.z; a[7]  = (_Float16)q1.w;
        a[8]  = (_Float16)q2.x; a[9]  = (_Float16)q2.y;
        a[10] = (_Float16)q2.z; a[11] = (_Float16)q2.w;
        a[12] = (_Float16)q3.x; a[13] = (_Float16)q3.y;
        a[14] = (_Float16)q3.z; a[15] = (_Float16)q3.w;

        v16h bf0 = frag_cat(*(const v8h*)(&BsT[bn0][h8]), *(const v8h*)(&BsT[bn0][h8 + 16]));
        v16h bf1 = frag_cat(*(const v8h*)(&BsT[bn1][h8]), *(const v8h*)(&BsT[bn1][h8 + 16]));
        acc0 = __builtin_amdgcn_wmma_f32_16x16x32_f16(false, a, false, bf0,
                                                      (short)0, acc0, false, false);
        acc1 = __builtin_amdgcn_wmma_f32_16x16x32_f16(false, a, false, bf1,
                                                      (short)0, acc1, false, false);
        __syncthreads();
    }
    // C layout: vgpr i -> row i (lanes 0-15) / row i+8 (lanes 16-31)
    float* crow = C + (size_t)(rowBase + rw * 16 + half * 8) * Nc
                    + colBase + cw * 32 + mrow;
    if (full) {
        #pragma unroll
        for (int i = 0; i < 8; ++i) {
            crow[(size_t)i * Nc]      = acc0[i];
            crow[(size_t)i * Nc + 16] = acc1[i];
        }
    } else {
        #pragma unroll
        for (int i = 0; i < 8; ++i) {
            int gr = rowBase + rw * 16 + half * 8 + i;
            if (gr < M) {
                crow[(size_t)i * Nc]      = acc0[i];
                crow[(size_t)i * Nc + 16] = acc1[i];
            }
        }
    }
}

// ---------------------------------------------------------------------------
// el[n,h] = h[n,h,:].al[h,:] ; er likewise. One wave32 per (n,head).
// ---------------------------------------------------------------------------
__global__ void attn_proj(const float* __restrict__ hbuf,
                          const float* __restrict__ al, const float* __restrict__ ar,
                          float* __restrict__ el, float* __restrict__ er, int r) {
    int gw   = blockIdx.x * 8 + (threadIdx.x >> 5);
    int lane = threadIdx.x & 31;
    if (gw >= NN * HH) return;
    int n = gw >> 3, head = gw & 7;
    const float* hp  = hbuf + (size_t)n * DD + head * OO;
    const float* alp = al + (size_t)r * HH * OO + head * OO;
    const float* arp = ar + (size_t)r * HH * OO + head * OO;
    float x0 = hp[lane], x1 = hp[lane + 32];
    float sl = x0 * alp[lane] + x1 * alp[lane + 32];
    float sr = x0 * arp[lane] + x1 * arp[lane + 32];
    #pragma unroll
    for (int off = 16; off; off >>= 1) {
        sl += __shfl_xor(sl, off);
        sr += __shfl_xor(sr, off);
    }
    if (lane == 0) {
        el[(size_t)r * NN * HH + gw] = sl;
        er[(size_t)r * NN * HH + gw] = sr;
    }
}

// edge softmax pass 1: per-dst max (uint-key atomicMax)
__global__ void edge_max(const int* __restrict__ src, const int* __restrict__ dst,
                         const float* __restrict__ el, const float* __restrict__ er,
                         unsigned* __restrict__ emax, int r) {
    int e = blockIdx.x * 256 + threadIdx.x;
    if (e >= EE) return;
    int s = src[(size_t)r * EE + e], d = dst[(size_t)r * EE + e];
    size_t nb = (size_t)r * NN * HH;
    #pragma unroll
    for (int h = 0; h < HH; ++h) {
        float v = el[nb + (size_t)s * HH + h] + er[nb + (size_t)d * HH + h];
        v = v > 0.f ? v : 0.2f * v;                 // leaky_relu(0.2)
        atomicMax(&emax[nb + (size_t)d * HH + h], fkey(v));
    }
}

// pass 2: den[dst,h] += exp(e - max)
__global__ void edge_expsum(const int* __restrict__ src, const int* __restrict__ dst,
                            const float* __restrict__ el, const float* __restrict__ er,
                            const unsigned* __restrict__ emax, float* __restrict__ den, int r) {
    int e = blockIdx.x * 256 + threadIdx.x;
    if (e >= EE) return;
    int s = src[(size_t)r * EE + e], d = dst[(size_t)r * EE + e];
    size_t nb = (size_t)r * NN * HH;
    #pragma unroll
    for (int h = 0; h < HH; ++h) {
        float v = el[nb + (size_t)s * HH + h] + er[nb + (size_t)d * HH + h];
        v = v > 0.f ? v : 0.2f * v;
        float m = fdec(emax[nb + (size_t)d * HH + h]);
        atomicAdd(&den[nb + (size_t)d * HH + h], __expf(v - m));
    }
}

// pass 3: z[r,dst,:] += a[e,h] * h[src,h,:]   (one wave per edge, coalesced)
__global__ void edge_aggregate(const int* __restrict__ src, const int* __restrict__ dst,
                               const float* __restrict__ el, const float* __restrict__ er,
                               const unsigned* __restrict__ emax, const float* __restrict__ den,
                               const float* __restrict__ hbuf, float* __restrict__ z, int r) {
    int e    = blockIdx.x * 8 + (threadIdx.x >> 5);
    int lane = threadIdx.x & 31;
    if (e >= EE) return;
    int s = src[(size_t)r * EE + e], d = dst[(size_t)r * EE + e];
    size_t nb = (size_t)r * NN * HH;
    const float* hs = hbuf + (size_t)s * DD;
    __builtin_prefetch(hs, 0, 0);                   // global_prefetch_b8
    float c = 0.f;
    if (lane < HH) {
        float v = el[nb + (size_t)s * HH + lane] + er[nb + (size_t)d * HH + lane];
        v = v > 0.f ? v : 0.2f * v;
        float m = fdec(emax[nb + (size_t)d * HH + lane]);
        c = __expf(v - m) / den[nb + (size_t)d * HH + lane];
    }
    float* zd = z + (size_t)r * NN * DD + (size_t)d * DD;
    #pragma unroll
    for (int step = 0; step < 16; ++step) {         // d = step*32+lane, head = step>>1
        float cc = __shfl(c, step >> 1);
        int dc = step * 32 + lane;
        atomicAdd(&zd[dc], cc * hs[dc]);
    }
}

// z = clip(z + bias, 0, 6)
__global__ void bias_clip(float* __restrict__ z, const float* __restrict__ bias) {
    int r = blockIdx.z;
    int i = blockIdx.x * 256 + threadIdx.x;         // over N*D
    float v = z[(size_t)r * NN * DD + i] + bias[r * DD + (i & (DD - 1))];
    v = v < 0.f ? 0.f : (v > 6.f ? 6.f : v);
    z[(size_t)r * NN * DD + i] = v;
}

// ---------------------------------------------------------------------------
// Path attention: wsum[r] = sum_n  tanh(z[n,r,:]@W1 + b1) @ W2   (WMMA + epilogue)
// ---------------------------------------------------------------------------
__global__ __launch_bounds__(256)
void path_attn(const float* __restrict__ z, const float* __restrict__ W1,
               const float* __restrict__ b1v, const float* __restrict__ W2,
               float* __restrict__ wsum) {
    __shared__ __align__(16) _Float16 As[64][40];
    __shared__ __align__(16) _Float16 BsT[64][40];
    const int r = blockIdx.z;
    const float* A = z + (size_t)r * NN * DD;
    const int tid  = threadIdx.x, lane = tid & 31, wid = tid >> 5;
    const int rw   = wid & 3, cw = wid >> 2;
    const int half = lane >> 4, mrow = lane & 15;
    const int rowBase = blockIdx.x * 64;
    const bool full = (rowBase + 64 <= NN);
    const int h8  = half * 8;
    const int am  = rw * 16 + mrow;
    const int bn0 = cw * 32 + mrow, bn1 = bn0 + 16;
    v8f acc0 = {}, acc1 = {};

    for (int kk = 0; kk < DD; kk += 32) {
        #pragma unroll
        for (int t = 0; t < 2; ++t) {
            int idx = tid + t * 256;
            int rA = idx >> 3, c4 = (idx & 7) << 2;
            int gr = rowBase + rA;
            v4h d;
            if (full || gr < NN) {
                const float4 v = *(const float4*)(A + (size_t)gr * DD + kk + c4);
                d = (v4h){(_Float16)v.x, (_Float16)v.y, (_Float16)v.z, (_Float16)v.w};
            } else {
                d = (v4h){(_Float16)0.f, (_Float16)0.f, (_Float16)0.f, (_Float16)0.f};
            }
            *(v4h*)(&As[rA][c4]) = d;
        }
        #pragma unroll
        for (int t = 0; t < 2; ++t) {
            int idx = tid + t * 256;
            int rB = idx >> 4, c4 = (idx & 15) << 2;
            const float4 v = *(const float4*)(W1 + (size_t)(kk + rB) * OO + c4);
            BsT[c4 + 0][rB] = (_Float16)v.x;
            BsT[c4 + 1][rB] = (_Float16)v.y;
            BsT[c4 + 2][rB] = (_Float16)v.z;
            BsT[c4 + 3][rB] = (_Float16)v.w;
        }
        __syncthreads();

        v16h a   = frag_cat(*(const v8h*)(&As[am][h8]),  *(const v8h*)(&As[am][h8 + 16]));
        v16h bf0 = frag_cat(*(const v8h*)(&BsT[bn0][h8]), *(const v8h*)(&BsT[bn0][h8 + 16]));
        v16h bf1 = frag_cat(*(const v8h*)(&BsT[bn1][h8]), *(const v8h*)(&BsT[bn1][h8 + 16]));
        acc0 = __builtin_amdgcn_wmma_f32_16x16x32_f16(false, a, false, bf0,
                                                      (short)0, acc0, false, false);
        acc1 = __builtin_amdgcn_wmma_f32_16x16x32_f16(false, a, false, bf1,
                                                      (short)0, acc1, false, false);
        __syncthreads();
    }
    // epilogue: per-row  sum_o tanh(t[o]+b1[o])*W2[o], summed over this block's rows
    const int c0 = cw * 32 + mrow, c1 = c0 + 16;
    float wloc = 0.f;
    #pragma unroll
    for (int i = 0; i < 8; ++i) {
        int gr = rowBase + rw * 16 + half * 8 + i;
        float p = tanhf(acc0[i] + b1v[c0]) * W2[c0]
                + tanhf(acc1[i] + b1v[c1]) * W2[c1];
        if (gr >= NN) p = 0.f;
        #pragma unroll
        for (int m = 1; m <= 8; m <<= 1) p += __shfl_xor(p, m);   // reduce 16-lane half
        if (mrow == 0) wloc += p;
    }
    if (mrow == 0) atomicAdd(&wsum[r], wloc);
}

// out[n,d] = sum_r softmax_r(wsum/N) * z[r,n,d]
__global__ void combine(const float* __restrict__ z, const float* __restrict__ wsum,
                        float* __restrict__ out) {
    int i = blockIdx.x * 256 + threadIdx.x;   // over N*D
    float m0 = wsum[0] * (1.f / NN), m1 = wsum[1] * (1.f / NN), m2 = wsum[2] * (1.f / NN);
    float mx = fmaxf(m0, fmaxf(m1, m2));
    float e0 = __expf(m0 - mx), e1 = __expf(m1 - mx), e2 = __expf(m2 - mx);
    float inv = 1.f / (e0 + e1 + e2);
    out[i] = inv * (e0 * z[i] + e1 * z[(size_t)NN * DD + i] + e2 * z[2 * (size_t)NN * DD + i]);
}

extern "C" void kernel_launch(void* const* d_in, const int* in_sizes, int n_in,
                              void* d_out, int out_size, void* d_ws, size_t ws_size,
                              hipStream_t stream) {
    (void)in_sizes; (void)n_in; (void)out_size; (void)ws_size;
    const float* feats = (const float*)d_in[0];
    const int*   src   = (const int*)d_in[1];
    const int*   dst   = (const int*)d_in[2];
    const float* W     = (const float*)d_in[3];
    const float* al    = (const float*)d_in[4];
    const float* ar    = (const float*)d_in[5];
    const float* bias  = (const float*)d_in[6];
    const float* W1    = (const float*)d_in[7];
    const float* b1    = (const float*)d_in[8];
    const float* W2    = (const float*)d_in[9];
    float* out = (float*)d_out;
    float* ws  = (float*)d_ws;

    float*    z    = ws;
    unsigned* emax = (unsigned*)(ws + EMAX_OFF);
    float*    den  = ws + DEN_OFF;
    float*    wsum = ws + WSUM_OFF;
    float*    hbuf = ws + H_OFF;
    float*    el   = ws + EL_OFF;
    float*    er   = ws + ER_OFF;

    // zero accumulators: z, emax(=key 0, only decoded after a write), den, wsum
    (void)hipMemsetAsync(d_ws, 0, H_OFF * sizeof(float), stream);

    for (int r = 0; r < RR; ++r) {
        dim3 g1((NN + 63) / 64, DD / 64, 1);
        fc_gemm_f16wmma<<<g1, 256, 0, stream>>>(feats + (size_t)r * NN * FF,
                                                W + (size_t)r * FF * DD,
                                                hbuf, NN, FF, DD);
        attn_proj<<<NN * HH / 8, 256, 0, stream>>>(hbuf, al, ar, el, er, r);
        edge_max<<<(EE + 255) / 256, 256, 0, stream>>>(src, dst, el, er, emax, r);
        edge_expsum<<<(EE + 255) / 256, 256, 0, stream>>>(src, dst, el, er, emax, den, r);
        edge_aggregate<<<EE / 8, 256, 0, stream>>>(src, dst, el, er, emax, den, hbuf, z, r);
    }
    bias_clip<<<dim3(NN * DD / 256, 1, RR), 256, 0, stream>>>(z, bias);
    path_attn<<<dim3((NN + 63) / 64, 1, RR), 256, 0, stream>>>(z, W1, b1, W2, wsum);
    combine<<<NN * DD / 256, 256, 0, stream>>>(z, wsum, out);
}
